// EnhancedAttentionV2_57019985822192
// MI455X (gfx1250) — compile-verified, matrix operand order checked by reading, not run on hardware
//
#include <hip/hip_runtime.h>
#include <hip/hip_bf16.h>

// Problem constants (match reference)
#define S_LEN    4096
#define HS       1024
#define NH       16
#define HD       64
#define WIN_HALF 128
#define LN_EPS   1e-12f

typedef __attribute__((ext_vector_type(16))) __bf16 v16bf;
typedef __attribute__((ext_vector_type(8)))  __bf16 v8bf;
typedef __attribute__((ext_vector_type(4)))  __bf16 v4bf;
typedef __attribute__((ext_vector_type(8)))  float  v8f;

// D = A(16x32 bf16) x B(32x16 bf16) + C(16x16 f32)  -> v_wmma_f32_16x16x32_bf16
__device__ __forceinline__ v8f wmma_bf16(v16bf a, v16bf b, v8f c) {
  return __builtin_amdgcn_wmma_f32_16x16x32_bf16(false, a, false, b, (short)0, c,
                                                 false, false);
}

// Load a 16-element bf16 fragment per the CDNA5 16-bit A/B VGPR layout:
// elements [0..7]  <- p[0..7]    (K base)       -- one B128 chunk
// elements [8..15] <- p[16..23]  (K base + 16)  -- one B128 chunk
// Caller passes p already offset by (lane>=16 ? 8 : 0).
__device__ __forceinline__ v16bf load_frag16(const __bf16* p) {
  v8bf lo = *(const v8bf*)p;
  v8bf hi = *(const v8bf*)(p + 16);
  v16bf r;
#pragma unroll
  for (int i = 0; i < 8; ++i) { r[i] = lo[i]; r[i + 8] = hi[i]; }
  return r;
}

// ---------------------------------------------------------------- f32 -> bf16
__global__ __launch_bounds__(256)
void cvt_kernel(const float* __restrict__ in, __bf16* __restrict__ out, int n4) {
  int i = blockIdx.x * 256 + threadIdx.x;
  if (i < n4) {
    float4 f = ((const float4*)in)[i];
    v4bf o;
    o[0] = (__bf16)f.x; o[1] = (__bf16)f.y;
    o[2] = (__bf16)f.z; o[3] = (__bf16)f.w;
    ((v4bf*)out)[i] = o;
  }
}

// ------------------------------------------------------------- QKV projection
// out[s,o] = sum_k X[s,k] * W[o,k] + bias[o]   (X @ W^T + b), bf16 in, f32 acc.
// transposed==0 : out is [S][HS] bf16 (Q, K)
// transposed==1 : out is [HS][S] bf16 (V^T, so attention reads V columns contiguously)
// Block = 8 waves; wave w owns rows [blk.x*256 + w*32, +32), cols [blk.y*64, +64).
// Inner K loop is ping-pong double-buffered (unroll 2) so fragment loads for the
// next K-step are in flight while the current 8 WMMAs execute.
__device__ __forceinline__ void load_ab(const __bf16* ap, const __bf16* bp, int k,
                                        v16bf a[2], v16bf b[4]) {
#pragma unroll
  for (int mc = 0; mc < 2; ++mc)
    a[mc] = load_frag16(ap + (size_t)mc * 16 * HS + k);
#pragma unroll
  for (int nc = 0; nc < 4; ++nc)
    b[nc] = load_frag16(bp + (size_t)nc * 16 * HS + k);
}

__global__ __launch_bounds__(256)
void qkv_gemm_kernel(const __bf16* __restrict__ Xb, const __bf16* __restrict__ Wb,
                     const float* __restrict__ bias, __bf16* __restrict__ outp,
                     int transposed) {
  const int lane = threadIdx.x;
  const int wave = threadIdx.y;
  const int hi = (lane >= 16) ? 1 : 0;
  const int lr = lane & 15;
  const int m0 = blockIdx.x * 256 + wave * 32;
  const int n0 = blockIdx.y * 64;

  v8f zero8 = {0.f, 0.f, 0.f, 0.f, 0.f, 0.f, 0.f, 0.f};
  v8f c[2][4] = {{zero8, zero8, zero8, zero8}, {zero8, zero8, zero8, zero8}};

  const __bf16* ap = Xb + (size_t)(m0 + lr) * HS + hi * 8;
  const __bf16* bp = Wb + (size_t)(n0 + lr) * HS + hi * 8;

  v16bf a0[2], b0[4], a1[2], b1[4];
  load_ab(ap, bp, 0, a0, b0);
  for (int k = 0; k < HS; k += 64) {
    load_ab(ap, bp, k + 32, a1, b1);             // prefetch odd step
#pragma unroll
    for (int mc = 0; mc < 2; ++mc)
#pragma unroll
      for (int nc = 0; nc < 4; ++nc)
        c[mc][nc] = wmma_bf16(a0[mc], b0[nc], c[mc][nc]);
    load_ab(ap, bp, (k + 64) & (HS - 1), a0, b0); // prefetch next even (wraps once)
#pragma unroll
    for (int mc = 0; mc < 2; ++mc)
#pragma unroll
      for (int nc = 0; nc < 4; ++nc)
        c[mc][nc] = wmma_bf16(a1[mc], b1[nc], c[mc][nc]);
  }

#pragma unroll
  for (int mc = 0; mc < 2; ++mc) {
#pragma unroll
    for (int nc = 0; nc < 4; ++nc) {
      int col = n0 + nc * 16 + lr;
      float bv = bias[col];
#pragma unroll
      for (int v = 0; v < 8; ++v) {
        int row = m0 + mc * 16 + v + hi * 8;  // C layout: VGPR v holds M=v (+8 hi)
        float val = c[mc][nc][v] + bv;
        if (!transposed) outp[(size_t)row * HS + col] = (__bf16)val;
        else             outp[(size_t)col * S_LEN + row] = (__bf16)val;
      }
    }
  }
}

// ------------------------------------------------- sliding-window attention
// One wave handles (head h, 16 query rows). Computes S^T = K*Q^T so that the
// exponentiated score fragment is directly the B operand of O^T = V^T * P^T
// (no cross-lane transpose). Softmax stats (m, l) are per-lane (lane&15 ==
// query), combined across lane pairs via one shfl_xor(16).
// All K and V fragment loads for a tile are issued up front so V-load latency
// overlaps the QK WMMAs and the exp/mask VALU work.
__global__ __launch_bounds__(256)
void attn_kernel(const __bf16* __restrict__ Qb, const __bf16* __restrict__ Kb,
                 const __bf16* __restrict__ Vt, const float* __restrict__ amask,
                 float* __restrict__ ctx) {
  const int lane = threadIdx.x;
  const int wave = threadIdx.y;
  const int task = blockIdx.x * 8 + wave;        // 0..4095
  const int h  = task & (NH - 1);
  const int q0 = (task >> 4) << 4;
  const int hi = (lane >= 16) ? 1 : 0;
  const int lr = lane & 15;

  // Q rows as B-fragments (lane = query, K = head dims)
  const __bf16* qp = Qb + (size_t)(q0 + lr) * HS + h * HD + hi * 8;
  v16bf qb0 = load_frag16(qp);        // dims 0..31
  v16bf qb1 = load_frag16(qp + 32);   // dims 32..63

  v8f zero8 = {0.f, 0.f, 0.f, 0.f, 0.f, 0.f, 0.f, 0.f};
  v8f ot[4] = {zero8, zero8, zero8, zero8};      // O^T, 4 dim-chunks of 16
  float m = -3.0e38f, l = 0.0f;

  int kt0 = q0 - WIN_HALF; if (kt0 < 0) kt0 = 0; kt0 &= ~31;
  int kend = q0 + 16 + WIN_HALF; if (kend > S_LEN) kend = S_LEN;
  const int i = q0 + lr;                         // this lane's query index

  for (int kt = kt0; kt < kend; kt += 32) {
    // ---- issue ALL fragment loads for this tile first (K then V)
    v16bf ka[2][2], va[4];
#pragma unroll
    for (int kc = 0; kc < 2; ++kc) {
      const __bf16* kp = Kb + (size_t)(kt + kc * 16 + lr) * HS + h * HD + hi * 8;
      ka[kc][0] = load_frag16(kp);
      ka[kc][1] = load_frag16(kp + 32);
    }
#pragma unroll
    for (int dc = 0; dc < 4; ++dc)
      va[dc] = load_frag16(Vt + (size_t)(h * HD + dc * 16 + lr) * S_LEN + kt + hi * 8);

    // ---- S^T tile: 32 keys x 16 queries (keys = WMMA M, dims = WMMA K)
    v8f st[2];
#pragma unroll
    for (int kc = 0; kc < 2; ++kc) {
      v8f s = zero8;
      s = wmma_bf16(ka[kc][0], qb0, s);
      s = wmma_bf16(ka[kc][1], qb1, s);
      st[kc] = s;
    }
    // ---- scale + padding mask + window mask (element: key j, query i)
    float tmax = -3.0e38f;
#pragma unroll
    for (int kc = 0; kc < 2; ++kc) {
      int jb = kt + kc * 16 + hi * 8;            // keys jb..jb+7 live in this lane
      float am[8];
      *(float4*)&am[0] = *(const float4*)(amask + jb);
      *(float4*)&am[4] = *(const float4*)(amask + jb + 4);
#pragma unroll
      for (int v = 0; v < 8; ++v) {
        int j = jb + v;
        float sv = st[kc][v] * 0.125f + (1.0f - am[v]) * (-10000.0f);
        bool keep = (j >= i - WIN_HALF) && (j < i + WIN_HALF);
        sv = keep ? sv : -10000.0f;
        st[kc][v] = sv;
        tmax = fmaxf(tmax, sv);
      }
    }
    // ---- online softmax (per-query stats live in lane pairs (q, q+16))
    tmax = fmaxf(tmax, __shfl_xor(tmax, 16, 32));
    float mn = fmaxf(m, tmax);
    float scale = __expf(m - mn);
    float rsum = 0.0f;
    v16bf pa;                                    // P^T as B-fragment for PV
#pragma unroll
    for (int v = 0; v < 8; ++v) {
      float p0 = __expf(st[0][v] - mn);
      float p1 = __expf(st[1][v] - mn);
      rsum += p0 + p1;
      pa[v]     = (__bf16)p0;                    // keys kt + v + hi*8
      pa[v + 8] = (__bf16)p1;                    // keys kt + 16 + v + hi*8
    }
    rsum += __shfl_xor(rsum, 16, 32);
    l = l * scale + rsum;
    m = mn;
    // ---- O^T += V^T_tile * P^T (A: M = dims, K = keys; contiguous in Vt)
#pragma unroll
    for (int dc = 0; dc < 4; ++dc) {
      v8f o = ot[dc];
#pragma unroll
      for (int v = 0; v < 8; ++v) o[v] *= scale; // rescale: lane's query matches stats
      ot[dc] = wmma_bf16(va[dc], pa, o);
    }
  }

  // ---- normalize & store ctx (f32). O^T: VGPR v = dim (v + hi*8), lane = query.
  float linv = 1.0f / l;
#pragma unroll
  for (int dc = 0; dc < 4; ++dc) {
    float* cp = ctx + (size_t)(q0 + lr) * HS + h * HD + dc * 16 + hi * 8;
    float4 o0 = { ot[dc][0] * linv, ot[dc][1] * linv, ot[dc][2] * linv, ot[dc][3] * linv };
    float4 o1 = { ot[dc][4] * linv, ot[dc][5] * linv, ot[dc][6] * linv, ot[dc][7] * linv };
    *(float4*)cp = o0;
    *(float4*)(cp + 4) = o1;
  }
}

// --------------------------------------------------- residual + layer norm
__global__ __launch_bounds__(256)
void ln_kernel(const float* __restrict__ hs, const float* __restrict__ ctx,
               const float* __restrict__ gamma, const float* __restrict__ beta,
               float* __restrict__ out) {
  __shared__ float red[8];
  const int row = blockIdx.x;
  const int t = threadIdx.x;
  const int lane = t & 31, wid = t >> 5;

  float4 hv = *(const float4*)(hs + (size_t)row * HS + t * 4);
  float4 cv = *(const float4*)(ctx + (size_t)row * HS + t * 4);
  float x[4] = {hv.x + cv.x, hv.y + cv.y, hv.z + cv.z, hv.w + cv.w};

  float s = x[0] + x[1] + x[2] + x[3];
#pragma unroll
  for (int o = 1; o < 32; o <<= 1) s += __shfl_xor(s, o, 32);
  if (lane == 0) red[wid] = s;
  __syncthreads();
  float tot = 0.f;
#pragma unroll
  for (int k = 0; k < 8; ++k) tot += red[k];
  float mu = tot * (1.0f / HS);

  float vs = 0.f;
#pragma unroll
  for (int k = 0; k < 4; ++k) { float d = x[k] - mu; vs += d * d; }
#pragma unroll
  for (int o = 1; o < 32; o <<= 1) vs += __shfl_xor(vs, o, 32);
  __syncthreads();
  if (lane == 0) red[wid] = vs;
  __syncthreads();
  float vtot = 0.f;
#pragma unroll
  for (int k = 0; k < 8; ++k) vtot += red[k];
  float rstd = rsqrtf(vtot * (1.0f / HS) + LN_EPS);

  float4 gv = *(const float4*)(gamma + t * 4);
  float4 bv = *(const float4*)(beta + t * 4);
  float4 ov;
  ov.x = (x[0] - mu) * rstd * gv.x + bv.x;
  ov.y = (x[1] - mu) * rstd * gv.y + bv.y;
  ov.z = (x[2] - mu) * rstd * gv.z + bv.z;
  ov.w = (x[3] - mu) * rstd * gv.w + bv.w;
  *(float4*)(out + (size_t)row * HS + t * 4) = ov;
}

// ----------------------------------------------------------------- launcher
extern "C" void kernel_launch(void* const* d_in, const int* in_sizes, int n_in,
                              void* d_out, int out_size, void* d_ws, size_t ws_size,
                              hipStream_t stream) {
  const float* hs    = (const float*)d_in[0];
  const float* amask = (const float*)d_in[1];
  const float* Wq    = (const float*)d_in[2];
  const float* bq    = (const float*)d_in[3];
  const float* Wk    = (const float*)d_in[4];
  const float* bk    = (const float*)d_in[5];
  const float* Wv    = (const float*)d_in[6];
  const float* bv    = (const float*)d_in[7];
  const float* gamma = (const float*)d_in[8];
  const float* beta  = (const float*)d_in[9];
  float* outp = (float*)d_out;

  // Workspace layout (~54 MB total; fits easily in MI455X's 192 MB L2)
  char* ws = (char*)d_ws;
  size_t off = 0;
  __bf16* Xb  = (__bf16*)(ws + off); off += (size_t)S_LEN * HS * 2;  // 8 MB
  __bf16* Wqb = (__bf16*)(ws + off); off += (size_t)HS * HS * 2;     // 2 MB
  __bf16* Wkb = (__bf16*)(ws + off); off += (size_t)HS * HS * 2;
  __bf16* Wvb = (__bf16*)(ws + off); off += (size_t)HS * HS * 2;
  __bf16* Qb  = (__bf16*)(ws + off); off += (size_t)S_LEN * HS * 2;  // [S][HS]
  __bf16* Kb  = (__bf16*)(ws + off); off += (size_t)S_LEN * HS * 2;  // [S][HS]
  __bf16* Vt  = (__bf16*)(ws + off); off += (size_t)S_LEN * HS * 2;  // [HS][S] (V^T)
  float*  ctx = (float*)(ws + off);  off += (size_t)S_LEN * HS * 4;  // 16 MB

  const int nX4 = S_LEN * HS / 4;  // 1048576
  const int nW4 = HS * HS / 4;     // 262144
  cvt_kernel<<<nX4 / 256, 256, 0, stream>>>(hs, Xb, nX4);
  cvt_kernel<<<nW4 / 256, 256, 0, stream>>>(Wq, Wqb, nW4);
  cvt_kernel<<<nW4 / 256, 256, 0, stream>>>(Wk, Wkb, nW4);
  cvt_kernel<<<nW4 / 256, 256, 0, stream>>>(Wv, Wvb, nW4);

  dim3 gg(S_LEN / 256, HS / 64), gb(32, 8);
  qkv_gemm_kernel<<<gg, gb, 0, stream>>>(Xb, Wqb, bq, Qb, 0);
  qkv_gemm_kernel<<<gg, gb, 0, stream>>>(Xb, Wkb, bk, Kb, 0);
  qkv_gemm_kernel<<<gg, gb, 0, stream>>>(Xb, Wvb, bv, Vt, 1);

  attn_kernel<<<(S_LEN / 16) * NH / 8, dim3(32, 8), 0, stream>>>(Qb, Kb, Vt, amask, ctx);
  ln_kernel<<<S_LEN, 256, 0, stream>>>(hs, ctx, gamma, beta, outp);
}
